// Model_84799834292719
// MI455X (gfx1250) — matrix operand order repaired and checked by reading, not compiled, not run on hardware
//
#include <hip/hip_runtime.h>
#include <hip/hip_bf16.h>

// ---------------------------------------------------------------------------
// Problem constants (from reference)
// ---------------------------------------------------------------------------
#define P_   2000
#define M_   150
#define DG_  2000
#define F_   256
#define H_   4
#define DH_  64
#define N1_  (P_ + M_)     // 2150
#define N2_  (P_ + DG_)    // 4000
#define SAH_ 128

typedef __attribute__((ext_vector_type(16))) _Float16 v16h;
typedef __attribute__((ext_vector_type(2)))  _Float16 v2h;
typedef __attribute__((ext_vector_type(8)))  float    v8f;

static inline int cdiv_i(long a, int b) { return (int)((a + b - 1) / b); }

// ---------------------------------------------------------------------------
// WMMA GEMM: C[Mr x Nc] = (RELUA? relu(A) : A)[Mr x K] @ B[K x Nc]
// fp32 in/out, fp16 WMMA compute, f32 accumulation. K must be multiple of 32.
// One wave per 16x32 output tile (2 accumulators, 2 v_wmma per K-step, shared
// A fragment). B tile staged TRANSPOSED in LDS as fp16: each lane's fragment
// is a contiguous 32B run (2x ds_load_b128), staging packs half-pairs
// (16x ds_store_b32 per lane). RELUA is a compile-time switch so the non-relu
// instantiation carries zero select overhead; relu lowers to v_max_f32.
// 4 waves / block, per-wave LDS region, no barriers needed (same-wave LDS
// ops are in-order on CDNA5).
// ---------------------------------------------------------------------------
template <int RELUA>
__global__ void __launch_bounds__(128)
gemm_wmma_f16(const float* __restrict__ A, int lda,
              const float* __restrict__ B, int ldb,
              float* __restrict__ C, int ldc,
              int Mr, int Nc, int K)
{
    // [wave][col 0..31][krow 0..31], krow contiguous -> 64B rows
    __shared__ __align__(32) _Float16 ldsBT[4][32][32];

    const int lane = threadIdx.x & 31;
    const int wave = threadIdx.x >> 5;
    const int mt  = (Mr + 15) >> 4;
    const int nt2 = (Nc + 31) >> 5;
    const int tiles = mt * nt2;
    const int tile = blockIdx.x * 4 + wave;
    if (tile >= tiles) return;                       // uniform per wave

    const int tm = tile % mt;
    const int tn = tile / mt;
    const int row0 = tm << 4;
    const int col0 = tn << 5;

    // A-fragment lane mapping used here (16-bit A 16x32, contiguous k order, same as B):
    //  lane<16 : row=lane,    K in {0..15}
    //  lane>=16: row=lane-16, K in {16..31}
    const int r    = lane & 15;
    const int koff = (lane >> 4) << 4;               // 0 or 16
    const int arow = row0 + r;
    const bool arowOK = (arow < Mr);

    // B-fragment lane mapping (B 32x16): lane<16 -> K 0..15 col=lane,
    // lane>=16 -> K 16..31 col=lane-16
    const int bk = (lane >> 4) << 4;                 // 0 or 16
    const int bc = lane & 15;

    // B staging: lane owns K-rows {2u, 2u+1} x 16 columns (cbase..cbase+15)
    const int u     = lane & 15;
    const int cbase = (lane >> 4) << 4;              // 0 or 16

    v8f acc0 = {};
    v8f acc1 = {};

    for (int k0 = 0; k0 < K; k0 += 32) {
        // ---- A fragment (direct global, fp32 -> fp16) ----
        v16h a = {};
        if (arowOK) {
            const float* ap = A + (long)arow * lda + k0 + koff;
            if (k0 + 32 < K)
                __builtin_prefetch(ap + 32, 0, 3);   // global_prefetch_b8
#pragma unroll
            for (int i = 0; i < 8; ++i) {
                float x0 = ap[i];
                float x1 = ap[8 + i];
                if (RELUA) { x0 = fmaxf(x0, 0.f); x1 = fmaxf(x1, 0.f); }
                a[i]     = (_Float16)x0;
                a[8 + i] = (_Float16)x1;
            }
        }

        // ---- stage B tile (32 K-rows x 32 cols) transposed in LDS ----
        {
            const int kr0 = k0 + 2 * u;
            const float* bp0 = B + (long)kr0 * ldb + col0 + cbase;
            const float* bp1 = bp0 + ldb;
            const bool ok0 = (kr0 < K);
            const bool ok1 = (kr0 + 1 < K);
#pragma unroll
            for (int j = 0; j < 16; ++j) {
                const int c = col0 + cbase + j;
                const bool cOK = (c < Nc);
                v2h p;
                p.x = (ok0 && cOK) ? (_Float16)bp0[j] : (_Float16)0.f;
                p.y = (ok1 && cOK) ? (_Float16)bp1[j] : (_Float16)0.f;
                *reinterpret_cast<v2h*>(&ldsBT[wave][cbase + j][2 * u]) = p;
            }
        }
        // gather two B fragments as contiguous 32B vectors
        const v16h b0 = *reinterpret_cast<const v16h*>(&ldsBT[wave][bc][bk]);
        const v16h b1 = *reinterpret_cast<const v16h*>(&ldsBT[wave][16 + bc][bk]);

        acc0 = __builtin_amdgcn_wmma_f32_16x16x32_f16(
            false, a, false, b0, (short)0, acc0, false, false);
        acc1 = __builtin_amdgcn_wmma_f32_16x16x32_f16(
            false, a, false, b1, (short)0, acc1, false, false);
    }

    // ---- store C: D VGPR v -> row row0 + v + 8*(lane>=16), col = lane&15 ----
    const int crow0 = row0 + ((lane >> 4) << 3);
    const int ccol0 = col0 + (lane & 15);
    const int ccol1 = ccol0 + 16;
#pragma unroll
    for (int v = 0; v < 8; ++v) {
        const int rr = crow0 + v;
        if (rr < Mr) {
            if (ccol0 < Nc) C[(long)rr * ldc + ccol0] = acc0[v];
            if (ccol1 < Nc) C[(long)rr * ldc + ccol1] = acc1[v];
        }
    }
}

// ---------------------------------------------------------------------------
// Sparse GCN: y[rows[e], :] += vals[e] * x[cols[e], :]  (atomic scatter)
// ---------------------------------------------------------------------------
__global__ void spmm_scatter(const int* __restrict__ rows, const int* __restrict__ cols,
                             const float* __restrict__ vals, const float* __restrict__ x,
                             float* __restrict__ y, long total /* = E * (F/4) */)
{
    long idx = (long)blockIdx.x * blockDim.x + threadIdx.x;
    if (idx >= total) return;
    const int e = (int)(idx >> 6);                   // F/4 = 64 chunks per edge
    const int c = ((int)idx & 63) << 2;
    const float v = vals[e];
    const float* xr = x + (long)cols[e] * F_ + c;
    float* yr = y + (long)rows[e] * F_ + c;
    atomicAdd(yr + 0, v * xr[0]);
    atomicAdd(yr + 1, v * xr[1]);
    atomicAdd(yr + 2, v * xr[2]);
    atomicAdd(yr + 3, v * xr[3]);
}

__global__ void leaky_acc(const float* __restrict__ tmp, float* __restrict__ lat,
                          float* __restrict__ acc, long n)
{
    long i = (long)blockIdx.x * blockDim.x + threadIdx.x;
    if (i >= n) return;
    float t = tmp[i];
    float v = t > 0.f ? t : 0.5f * t;
    lat[i] = v;
    acc[i] += v;
}

// ---------------------------------------------------------------------------
// GAT pieces
// ---------------------------------------------------------------------------
__global__ void gat_elr(const float* __restrict__ feat, const float* __restrict__ al,
                        const float* __restrict__ ar, float* __restrict__ el,
                        float* __restrict__ er, int n)
{
    int idx = blockIdx.x * blockDim.x + threadIdx.x;
    if (idx >= n * H_) return;
    const int node = idx >> 2, h = idx & 3;
    const float* f = feat + (long)node * F_ + h * DH_;
    const float* a = al + h * DH_;
    const float* b = ar + h * DH_;
    float sl = 0.f, sr = 0.f;
#pragma unroll 8
    for (int d = 0; d < DH_; ++d) { float fv = f[d]; sl += fv * a[d]; sr += fv * b[d]; }
    el[idx] = sl; er[idx] = sr;
}

__global__ void fill_f32(float* __restrict__ p, float v, long n)
{
    long i = (long)blockIdx.x * blockDim.x + threadIdx.x;
    if (i < n) p[i] = v;
}

__device__ inline void atomicMaxFloat(float* addr, float val)
{
    int* ai = reinterpret_cast<int*>(addr);
    int old = __float_as_int(*addr);
    while (__int_as_float(old) < val) {
        int prev = atomicCAS(ai, old, __float_as_int(val));
        if (prev == old) break;
        old = prev;
    }
}

__global__ void edge_max(const int* __restrict__ src, const int* __restrict__ dst,
                         const float* __restrict__ el, const float* __restrict__ er,
                         float* __restrict__ maxb, int E)
{
    int idx = blockIdx.x * blockDim.x + threadIdx.x;
    if (idx >= E * H_) return;
    const int e = idx >> 2, h = idx & 3;
    float v = el[src[e] * H_ + h] + er[dst[e] * H_ + h];
    v = v > 0.f ? v : 0.2f * v;                      // leaky_relu(0.2)
    atomicMaxFloat(&maxb[dst[e] * H_ + h], v);
}

__global__ void edge_exp(const int* __restrict__ src, const int* __restrict__ dst,
                         const float* __restrict__ el, const float* __restrict__ er,
                         const float* __restrict__ maxb, float* __restrict__ ex,
                         float* __restrict__ sumb, int E)
{
    int idx = blockIdx.x * blockDim.x + threadIdx.x;
    if (idx >= E * H_) return;
    const int e = idx >> 2, h = idx & 3;
    float v = el[src[e] * H_ + h] + er[dst[e] * H_ + h];
    v = v > 0.f ? v : 0.2f * v;
    float x = expf(v - maxb[dst[e] * H_ + h]);
    ex[idx] = x;
    atomicAdd(&sumb[dst[e] * H_ + h], x);
}

__global__ void edge_aggr(const int* __restrict__ src, const int* __restrict__ dst,
                          const float* __restrict__ ex, const float* __restrict__ sumb,
                          const float* __restrict__ feat, float* __restrict__ out, long total)
{
    long idx = (long)blockIdx.x * blockDim.x + threadIdx.x;
    if (idx >= total) return;                        // total = E * H * 16
    const int c = (int)idx & 15;
    const int h = ((int)(idx >> 4)) & 3;
    const int e = (int)(idx >> 6);
    const int s = src[e], d = dst[e];
    const float alpha = ex[e * H_ + h] / (sumb[d * H_ + h] + 1e-9f);
    const float* fp = feat + (long)s * F_ + h * DH_ + (c << 2);
    float* op = out + (long)d * F_ + h * DH_ + (c << 2);
    atomicAdd(op + 0, alpha * fp[0]);
    atomicAdd(op + 1, alpha * fp[1]);
    atomicAdd(op + 2, alpha * fp[2]);
    atomicAdd(op + 3, alpha * fp[3]);
}

__global__ void gat_finish(float* __restrict__ out, const float* __restrict__ b, long n)
{
    long i = (long)blockIdx.x * blockDim.x + threadIdx.x;
    if (i >= n) return;
    float x = out[i] + b[(int)i & (F_ - 1)];
    out[i] = x > 0.f ? x : (expf(x) - 1.f);          // elu
}

// ---------------------------------------------------------------------------
// Semantic attention
// ---------------------------------------------------------------------------
__global__ void sa_rowscore(const float* __restrict__ s, const float* __restrict__ b1,
                            const float* __restrict__ W2, float* __restrict__ wsum, int n)
{
    int row = blockIdx.x * blockDim.x + threadIdx.x;
    if (row >= n) return;
    const float* sp = s + (long)row * SAH_;
    float acc = 0.f;
#pragma unroll 8
    for (int j = 0; j < SAH_; ++j) acc += tanhf(sp[j] + b1[j]) * W2[j];
    atomicAdd(wsum, acc);
}

__global__ void sa_combine(const float* __restrict__ e0, const float* __restrict__ e1,
                           const float* __restrict__ wsum, float inv_n,
                           float* __restrict__ out, long n)
{
    long i = (long)blockIdx.x * blockDim.x + threadIdx.x;
    if (i >= n) return;
    const float w0 = wsum[0] * inv_n, w1 = wsum[1] * inv_n;
    const float m = fmaxf(w0, w1);
    const float x0 = expf(w0 - m), x1 = expf(w1 - m);
    const float b0 = x0 / (x0 + x1), b1 = x1 / (x0 + x1);
    out[i] = b0 * e0[i] + b1 * e1[i];
}

// ---------------------------------------------------------------------------
// Final scoring (collapsed): simi_pm[p,j] = M*(relu(patient)@Wt)[p,j]
//                                         + (sum_m relu(med[m]))@Wb[:,j] + M*out_b[j]
// ---------------------------------------------------------------------------
__global__ void med_colsum_relu(const float* __restrict__ med, float* __restrict__ svec)
{
    int f = blockIdx.x * blockDim.x + threadIdx.x;
    if (f >= F_) return;
    float s = 0.f;
    for (int m = 0; m < M_; ++m) { float v = med[(long)m * F_ + f]; s += fmaxf(v, 0.f); }
    svec[f] = s;
}

__global__ void bvec_kernel(const float* __restrict__ svec, const float* __restrict__ Wb,
                            float* __restrict__ bvec)
{
    int j = blockIdx.x * blockDim.x + threadIdx.x;
    if (j >= M_) return;
    float s = 0.f;
#pragma unroll 8
    for (int f = 0; f < F_; ++f) s += svec[f] * Wb[(long)f * M_ + j];
    bvec[j] = s;
}

__global__ void final_kernel(const float* __restrict__ Abuf, const float* __restrict__ bvec,
                             const float* __restrict__ out_b, float* __restrict__ simi, long n)
{
    long i = (long)blockIdx.x * blockDim.x + threadIdx.x;
    if (i >= n) return;
    const int j = (int)(i % M_);
    simi[i] = (float)M_ * Abuf[i] + bvec[j] + (float)M_ * out_b[j];
}

// ---------------------------------------------------------------------------
// Host orchestration
// ---------------------------------------------------------------------------
static void launch_gemm(const float* A, int lda, const float* B, int ldb,
                        float* C, int ldc, int Mr, int Nc, int K, int reluA,
                        hipStream_t s)
{
    const int mt = (Mr + 15) / 16, nt2 = (Nc + 31) / 32;
    const int blocks = (mt * nt2 + 3) / 4;
    if (reluA)
        gemm_wmma_f16<1><<<blocks, 128, 0, s>>>(A, lda, B, ldb, C, ldc, Mr, Nc, K);
    else
        gemm_wmma_f16<0><<<blocks, 128, 0, s>>>(A, lda, B, ldb, C, ldc, Mr, Nc, K);
}

extern "C" void kernel_launch(void* const* d_in, const int* in_sizes, int n_in,
                              void* d_out, int out_size, void* d_ws, size_t ws_size,
                              hipStream_t stream)
{
    // ---- inputs ----
    const int*   adj1_rows = (const int*)  d_in[0];
    const int*   adj1_cols = (const int*)  d_in[1];
    const float* adj1_vals = (const float*)d_in[2];
    const int*   adj2_rows = (const int*)  d_in[3];
    const int*   adj2_cols = (const int*)  d_in[4];
    const float* adj2_vals = (const float*)d_in[5];
    const int*   g_src[4] = { (const int*)d_in[6], (const int*)d_in[8],
                              (const int*)d_in[10], (const int*)d_in[12] };
    const int*   g_dst[4] = { (const int*)d_in[7], (const int*)d_in[9],
                              (const int*)d_in[11], (const int*)d_in[13] };
    const float* pEmbed = (const float*)d_in[15];
    const float* mEmbed = (const float*)d_in[16];
    const float* dEmbed = (const float*)d_in[17];
    const float* gat_W  = (const float*)d_in[18];
    const float* gat_al = (const float*)d_in[19];
    const float* gat_ar = (const float*)d_in[20];
    const float* gat_b  = (const float*)d_in[21];
    const float* sa_W1  = (const float*)d_in[22];
    const float* sa_b1  = (const float*)d_in[23];
    const float* sa_W2  = (const float*)d_in[24];
    const float* out_W  = (const float*)d_in[25];
    const float* out_b  = (const float*)d_in[26];

    const int E1  = in_sizes[0];
    const int E2  = in_sizes[3];
    const int EGP = in_sizes[6];
    const int EGM = in_sizes[10];
    const int gE[4] = { EGP, EGP, EGM, EGM };

    // ---- workspace layout (float offsets) ----
    float* ws = (float*)d_ws;
    float* lat1 = ws + 0;                 // N1*F = 550400
    float* tmp1 = ws + 550400;
    float* acc1 = ws + 1100800;
    float* lat2 = ws + 1651200;           // N2*F = 1024000
    float* tmp2 = ws + 2675200;
    float* acc2 = ws + 3699200;
    float* feat = ws + 4723200;           // P*F
    float* eb[4] = { ws + 5235200, ws + 5747200, ws + 6259200, ws + 6297600 };
    float* el   = ws + 6336000;           // n*H (<= 8000)
    float* er   = ws + 6344000;
    float* maxb = ws + 6352000;
    float* sumb = ws + 6360000;
    float* exb  = ws + 6368000;           // EGP*H
    float* s0   = ws + 6624000;           // P*SAH
    float* s1   = ws + 6880000;
    float* wsum = ws + 7136000;           // 2 (+pad)
    float* Abuf = ws + 7136008;           // P*M
    float* svec = ws + 7436008;           // F
    float* bvec = ws + 7436264;           // M

    // ---- outputs ----
    float* out = (float*)d_out;
    float* o_simi    = out + 0;           // P*M
    float* o_dgcn    = out + 300000;      // DG*F
    float* o_med     = out + 812000;      // M*F
    float* o_mgcn    = out + 850400;      // M*F
    float* o_patient = out + 888800;      // P*F

    const int T = 256;

    // ===== GCN over both graphs =====
    hipMemcpyAsync(lat1,            pEmbed, (size_t)P_ * F_ * 4, hipMemcpyDeviceToDevice, stream);
    hipMemcpyAsync(lat1 + P_ * F_,  mEmbed, (size_t)M_ * F_ * 4, hipMemcpyDeviceToDevice, stream);
    hipMemcpyAsync(lat2,            pEmbed, (size_t)P_ * F_ * 4, hipMemcpyDeviceToDevice, stream);
    hipMemcpyAsync(lat2 + P_ * F_,  dEmbed, (size_t)DG_ * F_ * 4, hipMemcpyDeviceToDevice, stream);
    hipMemsetAsync(acc1, 0, (size_t)N1_ * F_ * 4, stream);
    hipMemsetAsync(acc2, 0, (size_t)N2_ * F_ * 4, stream);

    for (int layer = 0; layer < 2; ++layer) {
        hipMemsetAsync(tmp1, 0, (size_t)N1_ * F_ * 4, stream);
        long t1 = (long)E1 * (F_ / 4);
        spmm_scatter<<<cdiv_i(t1, T), T, 0, stream>>>(adj1_rows, adj1_cols, adj1_vals, lat1, tmp1, t1);
        leaky_acc<<<cdiv_i((long)N1_ * F_, T), T, 0, stream>>>(tmp1, lat1, acc1, (long)N1_ * F_);

        hipMemsetAsync(tmp2, 0, (size_t)N2_ * F_ * 4, stream);
        long t2 = (long)E2 * (F_ / 4);
        spmm_scatter<<<cdiv_i(t2, T), T, 0, stream>>>(adj2_rows, adj2_cols, adj2_vals, lat2, tmp2, t2);
        leaky_acc<<<cdiv_i((long)N2_ * F_, T), T, 0, stream>>>(tmp2, lat2, acc2, (long)N2_ * F_);
    }

    // m_gcn / d_gcn outputs
    hipMemcpyAsync(o_mgcn, acc1 + (size_t)P_ * F_, (size_t)M_ * F_ * 4, hipMemcpyDeviceToDevice, stream);
    hipMemcpyAsync(o_dgcn, acc2 + (size_t)P_ * F_, (size_t)DG_ * F_ * 4, hipMemcpyDeviceToDevice, stream);

    // ===== 4 GAT layers =====
    const float* gat_h[4] = { acc1, acc2, acc1 + (size_t)P_ * F_, acc1 + (size_t)P_ * F_ };
    const int    gat_n[4] = { P_, P_, M_, M_ };

    for (int l = 0; l < 4; ++l) {
        const int n = gat_n[l];
        const int E = gE[l];
        // feat = h @ W_l  (n x 256) via WMMA
        launch_gemm(gat_h[l], F_, gat_W + (size_t)l * F_ * (H_ * DH_), H_ * DH_,
                    feat, H_ * DH_, n, H_ * DH_, F_, 0, stream);
        // per-node attention logits
        gat_elr<<<cdiv_i((long)n * H_, T), T, 0, stream>>>(
            feat, gat_al + l * H_ * DH_, gat_ar + l * H_ * DH_, el, er, n);
        // segment softmax over dst
        fill_f32<<<cdiv_i((long)n * H_, T), T, 0, stream>>>(maxb, -1e30f, (long)n * H_);
        hipMemsetAsync(sumb, 0, (size_t)n * H_ * 4, stream);
        edge_max<<<cdiv_i((long)E * H_, T), T, 0, stream>>>(g_src[l], g_dst[l], el, er, maxb, E);
        edge_exp<<<cdiv_i((long)E * H_, T), T, 0, stream>>>(g_src[l], g_dst[l], el, er, maxb, exb, sumb, E);
        // weighted aggregation
        hipMemsetAsync(eb[l], 0, (size_t)n * F_ * 4, stream);
        long ta = (long)E * H_ * 16;
        edge_aggr<<<cdiv_i(ta, T), T, 0, stream>>>(g_src[l], g_dst[l], exb, sumb, feat, eb[l], ta);
        gat_finish<<<cdiv_i((long)n * F_, T), T, 0, stream>>>(eb[l], gat_b + l * (H_ * DH_), (long)n * F_);
    }

    // ===== semantic attention: patient (e0,e1 over P), med (e2,e3 over M) =====
    // patient
    launch_gemm(eb[0], F_, sa_W1, SAH_, s0, SAH_, P_, SAH_, F_, 0, stream);
    launch_gemm(eb[1], F_, sa_W1, SAH_, s1, SAH_, P_, SAH_, F_, 0, stream);
    hipMemsetAsync(wsum, 0, 2 * 4, stream);
    sa_rowscore<<<cdiv_i(P_, T), T, 0, stream>>>(s0, sa_b1, sa_W2, wsum + 0, P_);
    sa_rowscore<<<cdiv_i(P_, T), T, 0, stream>>>(s1, sa_b1, sa_W2, wsum + 1, P_);
    sa_combine<<<cdiv_i((long)P_ * F_, T), T, 0, stream>>>(eb[0], eb[1], wsum, 1.f / P_,
                                                           o_patient, (long)P_ * F_);
    // med
    launch_gemm(eb[2], F_, sa_W1, SAH_, s0, SAH_, M_, SAH_, F_, 0, stream);
    launch_gemm(eb[3], F_, sa_W1, SAH_, s1, SAH_, M_, SAH_, F_, 0, stream);
    hipMemsetAsync(wsum, 0, 2 * 4, stream);
    sa_rowscore<<<cdiv_i(M_, T), T, 0, stream>>>(s0, sa_b1, sa_W2, wsum + 0, M_);
    sa_rowscore<<<cdiv_i(M_, T), T, 0, stream>>>(s1, sa_b1, sa_W2, wsum + 1, M_);
    sa_combine<<<cdiv_i((long)M_ * F_, T), T, 0, stream>>>(eb[2], eb[3], wsum, 1.f / M_,
                                                           o_med, (long)M_ * F_);

    // ===== collapsed final scoring =====
    // Abuf = relu(patient) @ out_W[:256,:]   (P x 150) via WMMA with fused relu on A
    launch_gemm(o_patient, F_, out_W, M_, Abuf, M_, P_, M_, F_, 1, stream);
    med_colsum_relu<<<1, F_, 0, stream>>>(o_med, svec);
    bvec_kernel<<<1, 256, 0, stream>>>(svec, out_W + (size_t)F_ * M_, bvec);
    final_kernel<<<cdiv_i((long)P_ * M_, T), T, 0, stream>>>(Abuf, bvec, out_b, o_simi, (long)P_ * M_);
}
